// FixedKAN_19078244728834
// MI455X (gfx1250) — matrix operand order repaired and checked by reading, not compiled, not run on hardware
//
#include <hip/hip_runtime.h>

// ---------------------------------------------------------------------------
// CDNA5 (gfx1250) KAN forward: bf16 WMMA GEMMs + fused Chebyshev epilogue.
// Double-buffered LDS staging; async global->LDS DMA when the toolchain
// exposes the gfx1250 async builtins (ASYNCcnt path), else b128 sync staging.
// ---------------------------------------------------------------------------

typedef __attribute__((ext_vector_type(16))) __bf16    v16bf;
typedef __attribute__((ext_vector_type(8)))  float     v8f;
typedef __attribute__((ext_vector_type(4)))  unsigned  u32x4;
typedef __attribute__((ext_vector_type(4)))  int       i32x4;

union FragU {
    u32x4 u[2];
    v16bf v;
};

#define BM 128
#define BN 128
#define BK 32
#define LDS_PAD 8            // 80-byte row stride -> conflict-free b128 LDS reads
#define BKP (BK + LDS_PAD)

#if __has_builtin(__builtin_amdgcn_global_load_async_to_lds_b128)
#  define HAVE_ASYNC 1
#else
#  define HAVE_ASYNC 0
#endif

#if __has_builtin(__builtin_amdgcn_s_wait_asynccnt)
#  define ASYNC_WAIT(n) __builtin_amdgcn_s_wait_asynccnt(n)
#else
#  define ASYNC_WAIT(n) asm volatile("s_wait_asynccnt %0" ::"i"(n))
#endif

#if HAVE_ASYNC
typedef __attribute__((address_space(1))) i32x4 g_i32x4;  // global (AS1)
typedef __attribute__((address_space(3))) i32x4 l_i32x4;  // LDS (AS3)
#endif

// ---------------------------------------------------------------------------
// f32 -> bf16 flat conversion (vectorized by 4)
// ---------------------------------------------------------------------------
__global__ void cvt_f32_to_bf16(const float* __restrict__ in,
                                __bf16* __restrict__ out, int n4) {
    int i = blockIdx.x * blockDim.x + threadIdx.x;
    if (i < n4) {
        float4 f = reinterpret_cast<const float4*>(in)[i];
        __bf16* o = out + 4 * i;
        o[0] = (__bf16)f.x; o[1] = (__bf16)f.y;
        o[2] = (__bf16)f.z; o[3] = (__bf16)f.w;
    }
}

// f32 [R][C] -> bf16 [C][R]  (transposing convert; small matrices)
__global__ void cvt_f32_to_bf16_t(const float* __restrict__ in,
                                  __bf16* __restrict__ out, int R, int C) {
    int i = blockIdx.x * blockDim.x + threadIdx.x;
    if (i < R * C) {
        int r = i / C, c = i - r * C;
        out[(size_t)c * R + r] = (__bf16)in[i];
    }
}

// ---------------------------------------------------------------------------
// Tiled bf16 WMMA GEMM:  D[M][N] = A[M][K] @ Bt[N][K]^T  (+ epilogue)
//   MODE 0: alpha = acc + bias; Chebyshev(coeffs) -> bf16 out
//   MODE 1: acc + bias -> bf16 out
//   MODE 2: acc + bias -> f32 out
// Block: 256 threads = 8 waves (4 x 2); wave tile 32x64 = 2x4 WMMA tiles.
// ---------------------------------------------------------------------------
template <int MODE>
__global__ void __launch_bounds__(256)
kan_gemm(const __bf16* __restrict__ A, const __bf16* __restrict__ Bt,
         const float* __restrict__ bias, const float* __restrict__ coeffs,
         __bf16* __restrict__ outb, float* __restrict__ outf,
         int M, int N, int K) {
    (void)M;
    __shared__ __bf16 As[2][BM][BKP];
    __shared__ __bf16 Bs[2][BN][BKP];

    const int tid  = threadIdx.x;
    const int lane = tid & 31;
    const int wid  = tid >> 5;
    const int wm   = wid >> 1;          // 0..3 -> 32-row band
    const int wn   = wid & 1;           // 0..1 -> 64-col band
    const int r    = lane & 15;
    const int half = lane >> 4;

    const int m0 = blockIdx.y * BM;
    const int n0 = blockIdx.x * BN;

    // cooperative staging: each thread moves 32B of A and 32B of B per tile
    const int lrow = tid >> 1;          // 0..127
    const int c0   = (tid & 1) * 16;    // element 0 or 16 within the 32-wide row

    const __bf16* Arow = A  + (size_t)(m0 + lrow) * K + c0;
    const __bf16* Brow = Bt + (size_t)(n0 + lrow) * K + c0;

    auto stage = [&](int buf, int k0) {
#if HAVE_ASYNC
        // ASYNCcnt-tracked DMA straight into LDS (4 x b128 per wave per stage)
        __builtin_amdgcn_global_load_async_to_lds_b128(
            (g_i32x4*)(const void*)(Arow + k0),
            (l_i32x4*)(void*)&As[buf][lrow][c0], 0, 0);
        __builtin_amdgcn_global_load_async_to_lds_b128(
            (g_i32x4*)(const void*)(Arow + k0 + 8),
            (l_i32x4*)(void*)&As[buf][lrow][c0 + 8], 0, 0);
        __builtin_amdgcn_global_load_async_to_lds_b128(
            (g_i32x4*)(const void*)(Brow + k0),
            (l_i32x4*)(void*)&Bs[buf][lrow][c0], 0, 0);
        __builtin_amdgcn_global_load_async_to_lds_b128(
            (g_i32x4*)(const void*)(Brow + k0 + 8),
            (l_i32x4*)(void*)&Bs[buf][lrow][c0 + 8], 0, 0);
#else
        u32x4 a0 = *reinterpret_cast<const u32x4*>(Arow + k0);
        u32x4 a1 = *reinterpret_cast<const u32x4*>(Arow + k0 + 8);
        u32x4 b0 = *reinterpret_cast<const u32x4*>(Brow + k0);
        u32x4 b1 = *reinterpret_cast<const u32x4*>(Brow + k0 + 8);
        *reinterpret_cast<u32x4*>(&As[buf][lrow][c0])     = a0;
        *reinterpret_cast<u32x4*>(&As[buf][lrow][c0 + 8]) = a1;
        *reinterpret_cast<u32x4*>(&Bs[buf][lrow][c0])     = b0;
        *reinterpret_cast<u32x4*>(&Bs[buf][lrow][c0 + 8]) = b1;
#endif
    };

    v8f acc[2][4] = {};

    const int NK = K / BK;
    stage(0, 0);                         // prologue: fill buffer 0

    for (int kt = 0; kt < NK; ++kt) {
        const int cur = kt & 1;
        if (kt + 1 < NK) {
            stage(cur ^ 1, (kt + 1) * BK);   // issue next tile into other buffer
#if HAVE_ASYNC
            ASYNC_WAIT(4);                   // current tile's 4 DMAs complete
#else
            if (kt + 2 < NK) {               // WGP-scope hint for tile after next
                __builtin_prefetch(Arow + (kt + 2) * BK, 0, 3);
                __builtin_prefetch(Brow + (kt + 2) * BK, 0, 3);
            }
#endif
        } else {
#if HAVE_ASYNC
            ASYNC_WAIT(0);
#endif
        }
        __syncthreads();                 // all waves' current tile visible

        // ---- fragment loads per ISA 7.12.2 lane layouts ----
        FragU af[2], bfr[4];
#pragma unroll
        for (int mt = 0; mt < 2; ++mt) {
            // A 16x32: elems 0..7 <- K[8h..8h+7], elems 8..15 <- K[8h+16..8h+23]
            const __bf16* p = &As[cur][wm * 32 + mt * 16 + r][half * 8];
            af[mt].u[0] = *reinterpret_cast<const u32x4*>(p);
            af[mt].u[1] = *reinterpret_cast<const u32x4*>(p + 16);
        }
#pragma unroll
        for (int nt = 0; nt < 4; ++nt) {
            // B 32x16: lane col = r, elems 0..7 <- K[8h..8h+7], elems 8..15 <- K[8h+16..8h+23]
            const __bf16* p = &Bs[cur][wn * 64 + nt * 16 + r][half * 8];
            bfr[nt].u[0] = *reinterpret_cast<const u32x4*>(p);
            bfr[nt].u[1] = *reinterpret_cast<const u32x4*>(p + 16);
        }
        __syncthreads();                 // reads drained; buffer reusable

#pragma unroll
        for (int mt = 0; mt < 2; ++mt)
#pragma unroll
            for (int nt = 0; nt < 4; ++nt)
                acc[mt][nt] = __builtin_amdgcn_wmma_f32_16x16x32_bf16(
                    false, af[mt].v, false, bfr[nt].v,
                    (short)0, acc[mt][nt], false, false);
    }

    // ---- epilogue: C/D layout -> (m = base + 8*half + j, n = base + r) ----
#pragma unroll
    for (int mt = 0; mt < 2; ++mt) {
        const int mbase = m0 + wm * 32 + mt * 16 + half * 8;
#pragma unroll
        for (int nt = 0; nt < 4; ++nt) {
            const int n  = n0 + wn * 64 + nt * 16 + r;
            const float bv = bias[n];
            float c[8];
            if constexpr (MODE == 0) {
                const float* cr = coeffs + (size_t)n * 8;
#pragma unroll
                for (int d = 0; d < 8; ++d) c[d] = cr[d];
            }
#pragma unroll
            for (int j = 0; j < 8; ++j) {
                float a = acc[mt][nt][j] + bv;
                float val;
                if constexpr (MODE == 0) {
                    float t0 = 1.0f, t1 = a;
                    float s = c[0] + c[1] * a;
#pragma unroll
                    for (int d = 2; d < 8; ++d) {
                        float t = 2.0f * a * t1 - t0;
                        s += c[d] * t;
                        t0 = t1; t1 = t;
                    }
                    val = s;
                } else {
                    val = a;
                }
                const size_t off = (size_t)(mbase + j) * N + n;
                if constexpr (MODE == 2) outf[off] = val;
                else                     outb[off] = (__bf16)val;
            }
        }
    }
}

// ---------------------------------------------------------------------------
// Host-side launcher
// ---------------------------------------------------------------------------
extern "C" void kernel_launch(void* const* d_in, const int* in_sizes, int n_in,
                              void* d_out, int out_size, void* d_ws, size_t ws_size,
                              hipStream_t stream) {
    (void)in_sizes; (void)n_in; (void)out_size; (void)ws_size;

    const int Mb = 16384, D0 = 1024, D1 = 1024, D2 = 256;

    const float* x       = (const float*)d_in[0];
    const float* w1      = (const float*)d_in[1];   // [D1][D0] == Bt layout
    const float* b1      = (const float*)d_in[2];
    const float* coeffs1 = (const float*)d_in[3];
    const float* cW1     = (const float*)d_in[4];   // [D1][D1], need transpose
    const float* cb1     = (const float*)d_in[5];
    const float* w2      = (const float*)d_in[6];   // [D2][D1] == Bt layout
    const float* b2      = (const float*)d_in[7];
    const float* coeffs2 = (const float*)d_in[8];
    const float* cW2     = (const float*)d_in[9];   // [D2][D2], need transpose
    const float* cb2     = (const float*)d_in[10];
    float* out = (float*)d_out;

    // carve bf16 workspace (all region sizes are multiples of 16 bytes)
    __bf16* ws = (__bf16*)d_ws;
    size_t o = 0;
    __bf16* xbf  = ws + o; o += (size_t)Mb * D0;
    __bf16* n1   = ws + o; o += (size_t)Mb * D1;
    __bf16* h1   = ws + o; o += (size_t)Mb * D1;
    __bf16* n2   = ws + o; o += (size_t)Mb * D2;
    __bf16* w1b  = ws + o; o += (size_t)D1 * D0;
    __bf16* cw1t = ws + o; o += (size_t)D1 * D1;
    __bf16* w2b  = ws + o; o += (size_t)D2 * D1;
    __bf16* cw2t = ws + o; o += (size_t)D2 * D2;

    const int T = 256;
    // conversions
    {
        int n4 = (Mb * D0) / 4;
        cvt_f32_to_bf16<<<(n4 + T - 1) / T, T, 0, stream>>>(x, xbf, n4);
        n4 = (D1 * D0) / 4;
        cvt_f32_to_bf16<<<(n4 + T - 1) / T, T, 0, stream>>>(w1, w1b, n4);
        n4 = (D2 * D1) / 4;
        cvt_f32_to_bf16<<<(n4 + T - 1) / T, T, 0, stream>>>(w2, w2b, n4);
        int ne = D1 * D1;
        cvt_f32_to_bf16_t<<<(ne + T - 1) / T, T, 0, stream>>>(cW1, cw1t, D1, D1);
        ne = D2 * D2;
        cvt_f32_to_bf16_t<<<(ne + T - 1) / T, T, 0, stream>>>(cW2, cw2t, D2, D2);
    }

    dim3 blk(256);
    // layer 1: alpha = x @ w1^T + b1 ; cheb(coeffs1) -> n1 (bf16)
    kan_gemm<0><<<dim3(D1 / BN, Mb / BM), blk, 0, stream>>>(
        xbf, w1b, b1, coeffs1, n1, nullptr, Mb, D1, D0);
    // layer 1 combine: h1 = n1 @ cW1 + cb1 -> bf16
    kan_gemm<1><<<dim3(D1 / BN, Mb / BM), blk, 0, stream>>>(
        n1, cw1t, cb1, nullptr, h1, nullptr, Mb, D1, D1);
    // layer 2: alpha = h1 @ w2^T + b2 ; cheb(coeffs2) -> n2 (bf16)
    kan_gemm<0><<<dim3(D2 / BN, Mb / BM), blk, 0, stream>>>(
        h1, w2b, b2, coeffs2, n2, nullptr, Mb, D2, D1);
    // layer 2 combine: out = n2 @ cW2 + cb2 -> f32
    kan_gemm<2><<<dim3(D2 / BN, Mb / BM), blk, 0, stream>>>(
        n2, cw2t, cb2, nullptr, nullptr, out, Mb, D2, D2);
}
